// LSTransformerEncoderLayer_46437186404363
// MI455X (gfx1250) — matrix operand order repaired and checked by reading, not compiled, not run on hardware
//
#include <hip/hip_runtime.h>

// MI455X / gfx1250 transformer encoder layer.
// Compute-bound (~223 GFLOP vs ~hundreds of MB of traffic) -> all five GEMM
// stages (QKV, scores, ctx, out-proj, FFN1/2) run on v_wmma_f32_16x16x32_f16
// with f32 accumulators. K-tile staging uses gfx1250 async global->LDS DMA
// (GLOBAL_LOAD_ASYNC_TO_LDS_B128, ASYNCcnt) with double-buffered LDS so the
// next tile streams in while the current tile's WMMAs execute.

typedef __attribute__((ext_vector_type(16))) _Float16 v16h;
typedef __attribute__((ext_vector_type(8)))  _Float16 v8h;
typedef __attribute__((ext_vector_type(8)))  float    v8f;

#define BM 128
#define BN 64
#define BK 32
#define LDL 40   // padded LDS row stride in halves (80B, 16B-aligned)

#if defined(__has_builtin)
#if __has_builtin(__builtin_amdgcn_global_load_async_to_lds_b128)
#define HAVE_ASYNC_LDS 1
#endif
#endif

// builtin signature (from clang diagnostic): GCC-style int __vector(4) pointers,
// AS1 (global/__device__) source, AS3 (LDS/__shared__) destination.
typedef int v4i __attribute__((vector_size(16)));
typedef __attribute__((address_space(1))) v4i* g128_t;
typedef __attribute__((address_space(3))) v4i* l128_t;

__device__ __forceinline__ void wait_async0() {
#if defined(__has_builtin) && __has_builtin(__builtin_amdgcn_s_wait_asynccnt)
  __builtin_amdgcn_s_wait_asynccnt(0);
#else
  asm volatile("s_wait_asynccnt 0x0" ::: "memory");
#endif
}

// 16-byte global -> LDS copy; async DMA on gfx1250 when available.
__device__ __forceinline__ void cp_async16(const _Float16* g, _Float16* l) {
#ifdef HAVE_ASYNC_LDS
  __builtin_amdgcn_global_load_async_to_lds_b128((g128_t)g, (l128_t)l, 0, 0);
#else
  *(v8h*)l = *(const v8h*)g;
#endif
}

__device__ __forceinline__ v8f wmma32(v16h a, v16h b, v8f c) {
  return __builtin_amdgcn_wmma_f32_16x16x32_f16(
      /*neg_a=*/false, a, /*neg_b=*/false, b,
      /*c_mod=*/(short)0, c, /*reuse_a=*/false, /*reuse_b=*/false);
}

// A fragment (16x32 f16): lane(low) = row; hi=0 -> K 0..7 & 16..23, hi=1 -> K 8..15 & 24..31
__device__ __forceinline__ v16h frag_a(const _Float16* t, int low, int hi) {
  const _Float16* p = t + low * LDL;
  v8h x0 = *(const v8h*)(p + hi * 8);
  v8h x1 = *(const v8h*)(p + 16 + hi * 8);
  v16h r;
#pragma unroll
  for (int i = 0; i < 8; ++i) { r[i] = x0[i]; r[i + 8] = x1[i]; }
  return r;
}

// B fragment (32x16 f16): lane(low) = column; hi=0 -> K 0..7 & 16..23, hi=1 -> K 8..15 & 24..31
__device__ __forceinline__ v16h frag_b(const _Float16* t, int low, int hi) {
  const _Float16* p = t + low * LDL;
  v8h x0 = *(const v8h*)(p + hi * 8);
  v8h x1 = *(const v8h*)(p + 16 + hi * 8);
  v16h r;
#pragma unroll
  for (int i = 0; i < 8; ++i) { r[i] = x0[i]; r[i + 8] = x1[i]; }
  return r;
}

// C[m,n] = sum_k A[m,k]*B[n,k]  (both row-major along K == NT GEMM)
// batched: z -> (zo = z/NH, zi = z%NH); base offsets per operand.
// epilogue: +bias[n], relu, +resid (same indexing as C), write f32 and/or f16.
__global__ __launch_bounds__(256)
void gemm_nt_kernel(const _Float16* __restrict__ A, long long a_outer, long long a_inner, int lda,
                    const _Float16* __restrict__ B, long long b_outer, long long b_inner, int ldb,
                    float* __restrict__ C32, _Float16* __restrict__ C16,
                    long long c_outer, long long c_inner, int ldc,
                    const float* __restrict__ bias, const float* __restrict__ resid,
                    int M, int N, int K, int NH, int do_relu)
{
  __shared__ _Float16 sA[2][BM * LDL];
  __shared__ _Float16 sB[2][BN * LDL];

  const int tid  = threadIdx.x;
  const int lane = tid & 31;
  const int low  = lane & 15;
  const int hi   = lane >> 4;
  const int wid  = tid >> 5;   // 0..7
  const int wm   = wid >> 1;   // 0..3  (M direction, 32 rows each)
  const int wn   = wid & 1;    // 0..1  (N direction, 32 cols each)

  const int zo = (int)blockIdx.z / NH;
  const int zi = (int)blockIdx.z % NH;
  const _Float16* Ab = A + (long long)zo * a_outer + (long long)zi * a_inner
                         + (long long)blockIdx.y * BM * lda;
  const _Float16* Bb = B + (long long)zo * b_outer + (long long)zi * b_inner
                         + (long long)blockIdx.x * BN * ldb;
  const long long c_off = (long long)zo * c_outer + (long long)zi * c_inner;

  v8f acc00 = {}; v8f acc01 = {}; v8f acc10 = {}; v8f acc11 = {};

  // per-thread staging coordinates: 16B chunks; A: rows rA and rA+64, B: row rA
  const int rA = tid >> 2;   // 0..63
  const int ch = tid & 3;    // 0..3 (chunk of 8 halves within BK=32)

  auto stage = [&](int buf, int k0) {
    const _Float16* ga = Ab + (long long)rA * lda + k0 + ch * 8;
    cp_async16(ga,                         &sA[buf][rA * LDL + ch * 8]);
    cp_async16(ga + (long long)64 * lda,   &sA[buf][(rA + 64) * LDL + ch * 8]);
    cp_async16(Bb + (long long)rA * ldb + k0 + ch * 8,
               &sB[buf][rA * LDL + ch * 8]);
  };

  const int nt = K / BK;
  stage(0, 0);                           // prologue: tile 0 -> buffer 0
  int buf = 0;
  for (int kt = 0; kt < nt; ++kt) {
    wait_async0();                       // our async copies for this tile done
    __syncthreads();                     // everyone's copies visible; prev reads drained
    if (kt + 1 < nt) {
      stage(buf ^ 1, (kt + 1) * BK);     // overlap next tile's DMA with compute
      if (kt + 2 < nt) {                 // keep L2 warm two tiles ahead
        __builtin_prefetch(Ab + (long long)(tid >> 1) * lda + (kt + 2) * BK, 0, 1);
        __builtin_prefetch(Bb + (long long)rA * ldb + (kt + 2) * BK, 0, 1);
      }
    }

    v16h a0 = frag_a(&sA[buf][(wm * 32 + 0)  * LDL], low, hi);
    v16h a1 = frag_a(&sA[buf][(wm * 32 + 16) * LDL], low, hi);
    v16h b0 = frag_b(&sB[buf][(wn * 32 + 0)  * LDL], low, hi);
    v16h b1 = frag_b(&sB[buf][(wn * 32 + 16) * LDL], low, hi);

    acc00 = wmma32(a0, b0, acc00);
    acc01 = wmma32(a0, b1, acc01);
    acc10 = wmma32(a1, b0, acc10);
    acc11 = wmma32(a1, b1, acc11);
    buf ^= 1;
  }

  const int rowBase = (int)blockIdx.y * BM + wm * 32;
  const int colBase = (int)blockIdx.x * BN + wn * 32;
#pragma unroll
  for (int mf = 0; mf < 2; ++mf) {
#pragma unroll
    for (int nf = 0; nf < 2; ++nf) {
      v8f acc = (mf == 0) ? (nf == 0 ? acc00 : acc01)
                          : (nf == 0 ? acc10 : acc11);
      const int col = colBase + nf * 16 + low;
      const float bv = bias ? bias[col] : 0.0f;
#pragma unroll
      for (int r = 0; r < 8; ++r) {
        const int row = rowBase + mf * 16 + r + hi * 8;
        float v = acc[r] + bv;
        if (do_relu) v = v > 0.0f ? v : 0.0f;
        const long long ci = c_off + (long long)row * ldc + col;
        if (resid) v += resid[ci];
        if (C32) C32[ci] = v;
        if (C16) C16[ci] = (_Float16)v;
      }
    }
  }
}

// LayerNorm over H=1024, one block (256 thr) per row, f16 output for WMMA A.
__global__ __launch_bounds__(256)
void ln_kernel(const float* __restrict__ x, const float* __restrict__ g,
               const float* __restrict__ b, _Float16* __restrict__ y)
{
  __shared__ float s1[256], s2[256];
  const long long row = blockIdx.x;
  const int tid = threadIdx.x;
  const float* xr = x + row * 1024;
  float4 v = ((const float4*)xr)[tid];
  float vv[4] = {v.x, v.y, v.z, v.w};
  float s = vv[0] + vv[1] + vv[2] + vv[3];
  float q = vv[0]*vv[0] + vv[1]*vv[1] + vv[2]*vv[2] + vv[3]*vv[3];
  s1[tid] = s; s2[tid] = q;
  __syncthreads();
  for (int st = 128; st > 0; st >>= 1) {
    if (tid < st) { s1[tid] += s1[tid + st]; s2[tid] += s2[tid + st]; }
    __syncthreads();
  }
  const float mu  = s1[0] * (1.0f / 1024.0f);
  const float var = s2[0] * (1.0f / 1024.0f) - mu * mu;
  const float inv = rsqrtf(var + 1e-5f);
  _Float16* yr = y + row * 1024;
#pragma unroll
  for (int j = 0; j < 4; ++j) {
    const int idx = tid * 4 + j;
    yr[idx] = (_Float16)((vv[j] - mu) * inv * g[idx] + b[idx]);
  }
}

// softmax over 512 keys: p = softmax(score*0.125 + mask[b,key]) -> f16
__global__ __launch_bounds__(128)
void softmax_kernel(const float* __restrict__ sc, const float* __restrict__ mask,
                    _Float16* __restrict__ P)
{
  __shared__ float red[128];
  const long long row = blockIdx.x;          // [0, 256*512)
  const int b = (int)(row >> 13);            // row / (NHEAD*S)
  const int tid = threadIdx.x;
  const float* r = sc + row * 512;
  const float* mrow = mask + (long long)b * 512;
  float vals[4];
  float mx = -3.4e38f;
#pragma unroll
  for (int j = 0; j < 4; ++j) {
    const int c = tid * 4 + j;
    vals[j] = r[c] * 0.125f + mrow[c];
    mx = fmaxf(mx, vals[j]);
  }
  red[tid] = mx; __syncthreads();
  for (int st = 64; st > 0; st >>= 1) {
    if (tid < st) red[tid] = fmaxf(red[tid], red[tid + st]);
    __syncthreads();
  }
  const float m = red[0];
  __syncthreads();
  float sum = 0.0f;
#pragma unroll
  for (int j = 0; j < 4; ++j) { vals[j] = __expf(vals[j] - m); sum += vals[j]; }
  red[tid] = sum; __syncthreads();
  for (int st = 64; st > 0; st >>= 1) {
    if (tid < st) red[tid] += red[tid + st];
    __syncthreads();
  }
  const float rinv = 1.0f / red[0];
  _Float16* pr = P + row * 512;
#pragma unroll
  for (int j = 0; j < 4; ++j) pr[tid * 4 + j] = (_Float16)(vals[j] * rinv);
}

// vT[(b*16+h)*64 + dd][s] = V[b, s, h, dd]  (so ctx GEMM is NT too)
__global__ __launch_bounds__(256)
void vtrans_kernel(const _Float16* __restrict__ qkv, _Float16* __restrict__ vT)
{
  const long long idx = (long long)blockIdx.x * 256 + threadIdx.x;  // < 256*64*512
  const int s  = (int)(idx & 511);
  const int dd = (int)((idx >> 9) & 63);
  const int z  = (int)(idx >> 15);
  const int b  = z >> 4, h = z & 15;
  vT[idx] = qkv[((long long)(b * 512 + s)) * 3072 + 2048 + h * 64 + dd];
}

__global__ __launch_bounds__(256)
void cvt_kernel(const float* __restrict__ src, _Float16* __restrict__ dst, int n)
{
  const int i = ((int)blockIdx.x * 256 + (int)threadIdx.x) * 4;
  if (i < n) {
    float4 v = *(const float4*)(src + i);
    dst[i]     = (_Float16)v.x;
    dst[i + 1] = (_Float16)v.y;
    dst[i + 2] = (_Float16)v.z;
    dst[i + 3] = (_Float16)v.w;
  }
}

extern "C" void kernel_launch(void* const* d_in, const int* in_sizes, int n_in,
                              void* d_out, int out_size, void* d_ws, size_t ws_size,
                              hipStream_t stream) {
  const float* x       = (const float*)d_in[0];
  const float* mask    = (const float*)d_in[1];
  const float* qkv_w   = (const float*)d_in[2];
  const float* qkv_b   = (const float*)d_in[3];
  const float* attn_ow = (const float*)d_in[4];
  const float* attn_ob = (const float*)d_in[5];
  const float* ln1_g   = (const float*)d_in[6];
  const float* ln1_b   = (const float*)d_in[7];
  const float* inter_w = (const float*)d_in[8];
  const float* inter_b = (const float*)d_in[9];
  const float* out_w   = (const float*)d_in[10];
  const float* out_b   = (const float*)d_in[11];
  const float* ln2_g   = (const float*)d_in[12];
  const float* ln2_b   = (const float*)d_in[13];
  float* out = (float*)d_out;

  const int Bn = 16, S = 512, H = 1024, NH = 16, IM = 4096;
  const int M = Bn * S;  // 8192
  const long long SS = (long long)S * S;

  // workspace carve-up (256B aligned)
  char* w = (char*)d_ws;
  auto carve = [&](size_t bytes) -> char* {
    char* p = w; w += (bytes + 255) & ~(size_t)255; return p;
  };
  _Float16* wqkv_h   = (_Float16*)carve((size_t)3 * H * H * 2);       // 6 MB
  _Float16* wattn_h  = (_Float16*)carve((size_t)H * H * 2);           // 2 MB
  _Float16* winter_h = (_Float16*)carve((size_t)IM * H * 2);          // 8 MB
  _Float16* wout_h   = (_Float16*)carve((size_t)H * IM * 2);          // 8 MB
  _Float16* y1_h     = (_Float16*)carve((size_t)M * H * 2);           // 16 MB
  _Float16* qkv_h    = (_Float16*)carve((size_t)M * 3 * H * 2);       // 48 MB
  _Float16* vT_h     = (_Float16*)carve((size_t)Bn * NH * 64 * S * 2);// 16 MB
  float*    scores   = (float*)   carve((size_t)Bn * NH * SS * 4);    // 256 MB
  _Float16* P_h      = (_Float16*)carve((size_t)Bn * NH * SS * 2);    // 128 MB
  _Float16* ctx_h    = (_Float16*)carve((size_t)M * H * 2);           // 16 MB
  float*    x1       = (float*)   carve((size_t)M * H * 4);           // 32 MB
  _Float16* y2_h     = (_Float16*)carve((size_t)M * H * 2);           // 16 MB
  _Float16* h_h      = (_Float16*)carve((size_t)M * IM * 2);          // 64 MB

  dim3 blk(256);

  // 1. weights -> f16
  cvt_kernel<<<(3 * H * H / 4 + 255) / 256, blk, 0, stream>>>(qkv_w,   wqkv_h,   3 * H * H);
  cvt_kernel<<<(H * H / 4 + 255) / 256,     blk, 0, stream>>>(attn_ow, wattn_h,  H * H);
  cvt_kernel<<<(IM * H / 4 + 255) / 256,    blk, 0, stream>>>(inter_w, winter_h, IM * H);
  cvt_kernel<<<(H * IM / 4 + 255) / 256,    blk, 0, stream>>>(out_w,   wout_h,   H * IM);

  // 2. LN1
  ln_kernel<<<M, blk, 0, stream>>>(x, ln1_g, ln1_b, y1_h);

  // 3. QKV GEMM: [8192,3072] = y1[8192,1024] x wqkv[3072,1024]^T (+bias) -> f16
  gemm_nt_kernel<<<dim3(3 * H / BN, M / BM, 1), blk, 0, stream>>>(
      y1_h, 0, 0, H, wqkv_h, 0, 0, H,
      nullptr, qkv_h, 0, 0, 3 * H, qkv_b, nullptr,
      M, 3 * H, H, 1, 0);

  // 4. V transpose per head
  vtrans_kernel<<<Bn * NH * 64 * S / 256, blk, 0, stream>>>(qkv_h, vT_h);

  // 5. scores = Q K^T  (256 batches of 512x512x64) -> f32
  gemm_nt_kernel<<<dim3(S / BN, S / BM, Bn * NH), blk, 0, stream>>>(
      qkv_h,            (long long)S * 3 * H, 64, 3 * H,
      qkv_h + H,        (long long)S * 3 * H, 64, 3 * H,
      scores, nullptr,  (long long)NH * SS, SS, S,
      nullptr, nullptr, S, S, 64, NH, 0);

  // 6. softmax(scale + mask) -> f16 probs
  softmax_kernel<<<Bn * NH * S, dim3(128), 0, stream>>>(scores, mask, P_h);

  // 7. ctx = P V  (256 batches of 512x64x512) -> f16 in [B,S,H] layout
  gemm_nt_kernel<<<dim3(64 / BN, S / BM, Bn * NH), blk, 0, stream>>>(
      P_h,   (long long)NH * SS, SS, S,
      vT_h,  (long long)NH * 64 * S, (long long)64 * S, S,
      nullptr, ctx_h, (long long)S * H, 64, H,
      nullptr, nullptr, S, 64, S, NH, 0);

  // 8. attn out proj + bias + residual(x) -> x1 (f32)
  gemm_nt_kernel<<<dim3(H / BN, M / BM, 1), blk, 0, stream>>>(
      ctx_h, 0, 0, H, wattn_h, 0, 0, H,
      x1, nullptr, 0, 0, H, attn_ob, x,
      M, H, H, 1, 0);

  // 9. LN2
  ln_kernel<<<M, blk, 0, stream>>>(x1, ln2_g, ln2_b, y2_h);

  // 10. FFN1 + bias + relu -> f16
  gemm_nt_kernel<<<dim3(IM / BN, M / BM, 1), blk, 0, stream>>>(
      y2_h, 0, 0, H, winter_h, 0, 0, H,
      nullptr, h_h, 0, 0, IM, inter_b, nullptr,
      M, IM, H, 1, 1);

  // 11. FFN2 + bias + residual(x1) -> d_out (f32)
  gemm_nt_kernel<<<dim3(H / BN, M / BM, 1), blk, 0, stream>>>(
      h_h, 0, 0, IM, wout_h, 0, 0, IM,
      out, nullptr, 0, 0, H, out_b, x1,
      M, H, IM, 1, 0);

  (void)in_sizes; (void)n_in; (void)out_size; (void)ws_size;
}
